// NeRF_24945170055548
// MI455X (gfx1250) — matrix operand order repaired and checked by reading, not compiled, not run on hardware
//
#include <hip/hip_runtime.h>
#include <hip/hip_bf16.h>
#include <stdint.h>

// ---------------------------------------------------------------------------
// Fused NeRF head for MI455X (gfx1250, wave32, WMMA bf16 16x16x32, f32 acc).
// Kernel 1: pack all weights into WMMA-B fragment-major bf16 in d_ws.
// Kernel 2: precompute attention K/V for the 4 parts + bias/LN tables.
// Kernel 3: fused megakernel; each wave processes a 16-point tile through
//           feat-branch, index-branch, 4 rgbnets, softmax blend, sigmoid.
//           Stage weights are staged global->LDS with CDNA5 async copies
//           (GLOBAL_LOAD_ASYNC_TO_LDS_B128 + s_wait_asynccnt).
// ---------------------------------------------------------------------------

typedef __attribute__((ext_vector_type(16))) __bf16 v16bf;
typedef __attribute__((ext_vector_type(8)))  float  v8f;

#define N_PTS   262144
#define WAVES   4

// ---- packed-weight fragment bases (units of 512 bf16 = one 32x16 fragment)
#define FB_FEAT_W1    0      // 64x128  -> 16 frags (K padded 39->64)
#define FB_FEAT_WQ    16     // 128x32  -> 8
#define FB_FEAT_FFN1  24     // 128x256 -> 64 (n-major: N-halves contiguous)
#define FB_FEAT_FFN2  88     // 256x128 -> 64 (k-major: K-halves contiguous)
#define FB_FEAT_WOUT  152    // 128x128 -> 32
#define FB_IDX_W1     184    // 32x128  -> 8 (K padded 12->32)
#define FB_IDX_WQ     192
#define FB_IDX_FFN1   200
#define FB_IDX_FFN2   264
#define FB_IDX_WOUT   328    // 128x16 (N padded 5->16) -> 8
#define FB_RGB0       336    // per part: l0 32, l1 32, l2 8  (stride 72)
#define NFRAG_TOTAL   624

// ---- small fp32 table offsets (floats), placed after fragments in d_ws
#define SF_FEAT_B1     0
#define SF_FEAT_BQ     128
#define SF_FEAT_FFN1B  160
#define SF_FEAT_FFN2B  416
#define SF_FEAT_LN1G   544
#define SF_FEAT_LN1B   672
#define SF_FEAT_LN2G   800
#define SF_FEAT_LN2B   928
#define SF_FEAT_OUTB   1056
#define SF_FEAT_K4     1184
#define SF_FEAT_V4     1312
#define SF_IDX_B1      1824
#define SF_IDX_BQ      1952
#define SF_IDX_FFN1B   1984
#define SF_IDX_FFN2B   2240
#define SF_IDX_LN1G    2368
#define SF_IDX_LN1B    2496
#define SF_IDX_LN2G    2624
#define SF_IDX_LN2B    2752
#define SF_IDX_OUTB    2880
#define SF_IDX_K4      2896
#define SF_IDX_V4      3024
#define SF_RGB_B0      3536   // per part: b0[128] b1[128] b2[16] (stride 272)
#define SF_TOTAL       4624

struct Params {
    const float* p[64];        // flattened params (d_in[4..67], insertion order)
    const float* part_shape;   // d_in[2]
    const float* part_tex;     // d_in[3]
};

// ---------------------------------------------------------------------------
// helpers
// ---------------------------------------------------------------------------
__device__ inline v8f vzero8() {
    v8f c;
#pragma unroll
    for (int j = 0; j < 8; ++j) c[j] = 0.0f;
    return c;
}

// A fragment (16x32 bf16) from LDS row-major [16][LDK]; ISA A-layout:
// lane = M row, vgpr i holds K pair; K = (i/4)*16 + (lane/16)*8 + (i%4)*2.
template<int LDK>
__device__ inline v16bf load_A(const __bf16* buf, int kt) {
    int lane = threadIdx.x & 31;
    int hi = lane >> 4, m = lane & 15;
    const __bf16* row = buf + m * LDK + kt * 32;
    union { v16bf v; unsigned int u[8]; } A;
#pragma unroll
    for (int i = 0; i < 8; ++i) {
        int k = ((i >> 2) << 4) + (hi << 3) + ((i & 3) << 1);
        A.u[i] = *(const unsigned int*)(row + k);
    }
    return A.v;
}

// B fragment: pre-packed lane-major, 32 lanes x 16 bf16 contiguous (1KB).
__device__ inline v16bf load_B(const __bf16* wb, int frag) {
    int lane = threadIdx.x & 31;
    return *(const v16bf*)(wb + frag * 512 + lane * 16);
}

// accumulate C[nt] += A(16xKT*32) * B over the staged fragments
template<int LDK, int KT, int NT, int KM>
__device__ inline void mm_acc(const __bf16* in, int kt0, v8f* C, const __bf16* wb) {
    v16bf A[KT];
#pragma unroll
    for (int kt = 0; kt < KT; ++kt) A[kt] = load_A<LDK>(in, kt0 + kt);
#pragma unroll
    for (int nt = 0; nt < NT; ++nt) {
#pragma unroll
        for (int kt = 0; kt < KT; ++kt) {
            int f = KM ? (kt * NT + nt) : (nt * KT + kt);
            C[nt] = __builtin_amdgcn_wmma_f32_16x16x32_bf16(
                false, A[kt], false, load_B(wb, f), (short)0, C[nt], false, false);
        }
    }
}

// full stage: matmul + bias (+relu) -> bf16 LDS buffer (C layout: N=lane, M=vgpr)
template<int LDK, int KT, int NT, int KM, int LDN, int RELU>
__device__ inline void mm_stage(const __bf16* in, __bf16* outp, int ncol0,
                                const __bf16* wb, const float* bias) {
    v8f C[NT];
#pragma unroll
    for (int i = 0; i < NT; ++i) C[i] = vzero8();
    mm_acc<LDK, KT, NT, KM>(in, 0, C, wb);
    int lane = threadIdx.x & 31, cn = lane & 15, hi = lane >> 4;
#pragma unroll
    for (int nt = 0; nt < NT; ++nt) {
        float bb = bias[nt * 16 + cn];
#pragma unroll
        for (int r = 0; r < 8; ++r) {
            float v = C[nt][r] + bb;
            if (RELU) v = fmaxf(v, 0.0f);
            outp[(r + (hi << 3)) * LDN + ncol0 + nt * 16 + cn] = (__bf16)v;
        }
    }
}

// cooperative global->LDS weight staging using the CDNA5 async DMA path.
// Each lane issues 16-byte async copies (one instruction moves 512B/wave),
// then waits on its own ASYNCcnt; the callers' __syncthreads() then makes
// the buffer visible to every wave in the block.
__device__ inline void copy_w(__bf16* lds, const __bf16* g, int halves) {
    int n = halves >> 3;                           // number of 16B chunks
    unsigned lbase = (unsigned)(uintptr_t)lds;     // LDS offset (low 32 bits)
    for (int i = threadIdx.x; i < n; i += blockDim.x) {
        unsigned laddr = lbase + (unsigned)i * 16u;
        const void* gaddr = (const char*)g + (size_t)i * 16u;
        asm volatile("global_load_async_to_lds_b128 %0, %1, off"
                     :: "v"(laddr), "v"(gaddr)
                     : "memory");
    }
    asm volatile("s_wait_asynccnt 0x0" ::: "memory");
}

// per-point layernorm over 128 channels; partner lanes (l, l^16) split columns
__device__ inline void ln_wave(__bf16* X, const float* g, const float* b) {
    int lane = threadIdx.x & 31, m = lane & 15, hi = lane >> 4;
    float s = 0.f, s2 = 0.f;
    for (int c = hi * 64; c < hi * 64 + 64; ++c) {
        float v = (float)X[m * 128 + c];
        s += v; s2 += v * v;
    }
    s  += __shfl_xor(s, 16);
    s2 += __shfl_xor(s2, 16);
    float mu = s * (1.0f / 128.0f);
    float var = s2 * (1.0f / 128.0f) - mu * mu;
    float r = rsqrtf(var + 1e-5f);
    for (int c = hi * 64; c < hi * 64 + 64; ++c) {
        float v = (float)X[m * 128 + c];
        X[m * 128 + c] = (__bf16)((v - mu) * r * g[c] + b[c]);
    }
}

// cross attention against 4 precomputed part K/V; lane = (point, 2 heads)
__device__ inline void attention(__bf16* X, const float* Q,
                                 const float* K4, const float* V4) {
    int lane = threadIdx.x & 31, m = lane & 15, hi = lane >> 4;
    const float scale = 0.17677669529663687f;   // 32^-0.5 (full dim_head)
#pragma unroll
    for (int hh = 0; hh < 2; ++hh) {
        int h = hi * 2 + hh;
        float sc[4]; float mx = -3.0e38f;
#pragma unroll
        for (int p = 0; p < 4; ++p) {
            float s = 0.f;
#pragma unroll
            for (int d = 0; d < 8; ++d)
                s += Q[m * 32 + h * 8 + d] * K4[p * 32 + h * 8 + d];
            sc[p] = s * scale; mx = fmaxf(mx, sc[p]);
        }
        float den = 0.f;
#pragma unroll
        for (int p = 0; p < 4; ++p) { sc[p] = expf(sc[p] - mx); den += sc[p]; }
        float inv = 1.0f / den;
        for (int d = 0; d < 32; ++d) {
            float o = 0.f;
#pragma unroll
            for (int p = 0; p < 4; ++p) o += sc[p] * V4[p * 128 + h * 32 + d];
            int c = h * 32 + d;
            X[m * 128 + c] = (__bf16)((float)X[m * 128 + c] + o * inv);
        }
    }
}

struct BC {
    int fb_w1, fb_wq, fb_ffn1, fb_ffn2;
    int sf_b1, sf_bq, sf_ffn1b, sf_ffn2b;
    int sf_ln1g, sf_ln1b, sf_ln2g, sf_ln2b;
    int sf_k4, sf_v4;
};

// shared body of both _index_mlp branches (ends with relu(x2) in X)
template<int KTIN>
__device__ void run_branch(__bf16* X, __bf16* H, float* Q,
                           const __bf16* frags, const float* sm,
                           __bf16* wbuf, const BC bc) {
    int lane = threadIdx.x & 31, cn = lane & 15, hi = lane >> 4, m = cn;
    // S1: H (16 x KTIN*32) -> X (16x128)
    copy_w(wbuf, frags + bc.fb_w1 * 512, KTIN * 8 * 512);
    __syncthreads();
    mm_stage<KTIN * 32, KTIN, 8, 1, 128, 0>(H, X, 0, wbuf, sm + bc.sf_b1);
    __syncthreads();
    // S2: q = X @ Wq  (fp32 into Q)
    copy_w(wbuf, frags + bc.fb_wq * 512, 8 * 512);
    __syncthreads();
    {
        v8f C[2]; C[0] = vzero8(); C[1] = vzero8();
        mm_acc<128, 4, 2, 1>(X, 0, C, wbuf);
        const float* bq = sm + bc.sf_bq;
#pragma unroll
        for (int nt = 0; nt < 2; ++nt)
#pragma unroll
            for (int r = 0; r < 8; ++r)
                Q[(r + hi * 8) * 32 + nt * 16 + cn] = C[nt][r] + bq[nt * 16 + cn];
    }
    __syncthreads();
    attention(X, Q, sm + bc.sf_k4, sm + bc.sf_v4);
    __syncthreads();
    ln_wave(X, sm + bc.sf_ln1g, sm + bc.sf_ln1b);
    __syncthreads();
    // FFN1 (128->256) in two N-halves, relu, into H (16x256)
    copy_w(wbuf, frags + bc.fb_ffn1 * 512, 32 * 512);
    __syncthreads();
    mm_stage<128, 4, 8, 0, 256, 1>(X, H, 0, wbuf, sm + bc.sf_ffn1b);
    __syncthreads();
    copy_w(wbuf, frags + (bc.fb_ffn1 + 32) * 512, 32 * 512);
    __syncthreads();
    mm_stage<128, 4, 8, 0, 256, 1>(X, H, 128, wbuf, sm + bc.sf_ffn1b + 128);
    __syncthreads();
    // FFN2 (256->128) in two K-halves, accumulate in regs; + bias + residual
    {
        v8f C[8];
#pragma unroll
        for (int i = 0; i < 8; ++i) C[i] = vzero8();
        copy_w(wbuf, frags + bc.fb_ffn2 * 512, 32 * 512);
        __syncthreads();
        mm_acc<256, 4, 8, 1>(H, 0, C, wbuf);
        __syncthreads();
        copy_w(wbuf, frags + (bc.fb_ffn2 + 32) * 512, 32 * 512);
        __syncthreads();
        mm_acc<256, 4, 8, 1>(H, 4, C, wbuf);
        const float* b2 = sm + bc.sf_ffn2b;
#pragma unroll
        for (int nt = 0; nt < 8; ++nt)
#pragma unroll
            for (int r = 0; r < 8; ++r) {
                int mr = r + hi * 8, c = nt * 16 + cn;
                X[mr * 128 + c] =
                    (__bf16)((float)X[mr * 128 + c] + C[nt][r] + b2[c]);
            }
    }
    __syncthreads();
    ln_wave(X, sm + bc.sf_ln2g, sm + bc.sf_ln2b);
    __syncthreads();
    // relu(x2) in place (input of the 'out' linear)
    for (int c = hi * 64; c < hi * 64 + 64; ++c) {
        float v = (float)X[m * 128 + c];
        X[m * 128 + c] = (__bf16)fmaxf(v, 0.0f);
    }
    __syncthreads();
}

// ---------------------------------------------------------------------------
// Kernel 1: pack one 32x16 fragment per block into d_ws (fragment-major bf16)
// ---------------------------------------------------------------------------
__global__ __launch_bounds__(256) void pack_kernel(Params P, __bf16* frags) {
    const int fb[23] = {0,16,24,88,152,184,192,200,264,328,
                        336,368,400,408,440,472,480,512,544,552,584,616,624};
    const int pidx[22] = {0,4,10,12,18, 20,24,30,32,38,
                          40,42,44, 46,48,50, 52,54,56, 58,60,62};
    const int Ks[22] = {39,128,128,256,128, 12,128,128,256,128,
                        128,128,128, 128,128,128, 128,128,128, 128,128,128};
    const int Ns[22] = {128,32,256,128,128, 128,32,256,128,5,
                        128,128,3, 128,128,3, 128,128,3, 128,128,3};
    const int km[22] = {1,1,0,1,1, 1,1,0,1,1, 1,1,1, 1,1,1, 1,1,1, 1,1,1};
    int b = blockIdx.x;
    int s = 0;
    for (int i = 0; i < 22; ++i) if (b >= fb[i]) s = i;
    int lf = b - fb[s];
    int KT = (Ks[s] + 31) / 32, NT = (Ns[s] + 15) / 16;
    int kt, nt;
    if (km[s]) { kt = lf / NT; nt = lf % NT; }
    else       { nt = lf / KT; kt = lf % KT; }
    const float* W = P.p[pidx[s]];
    int Kr = Ks[s], Nr = Ns[s];
    __bf16* dst = frags + b * 512;
    for (int e = threadIdx.x; e < 512; e += blockDim.x) {
        int lane = e >> 4, i = e & 15;
        int k = kt * 32 + ((i >> 3) << 4) + ((lane >> 4) << 3) + (i & 7);   // K within padded tile
        int col = nt * 16 + (lane & 15);
        float v = (k < Kr && col < Nr) ? W[k * Nr + col] : 0.0f;
        dst[e] = (__bf16)v;
    }
}

// ---------------------------------------------------------------------------
// Kernel 2: bias/LN tables + part-side attention K/V precompute (fp32)
// ---------------------------------------------------------------------------
__global__ __launch_bounds__(256) void prep2_kernel(Params P, float* sm) {
    struct CE { short pi; short n; short dst; short pad; };
    const CE tab[30] = {
        {1,128,SF_FEAT_B1,128},{5,32,SF_FEAT_BQ,32},
        {11,256,SF_FEAT_FFN1B,256},{13,128,SF_FEAT_FFN2B,128},
        {14,128,SF_FEAT_LN1G,128},{15,128,SF_FEAT_LN1B,128},
        {16,128,SF_FEAT_LN2G,128},{17,128,SF_FEAT_LN2B,128},
        {19,128,SF_FEAT_OUTB,128},
        {21,128,SF_IDX_B1,128},{25,32,SF_IDX_BQ,32},
        {31,256,SF_IDX_FFN1B,256},{33,128,SF_IDX_FFN2B,128},
        {34,128,SF_IDX_LN1G,128},{35,128,SF_IDX_LN1B,128},
        {36,128,SF_IDX_LN2G,128},{37,128,SF_IDX_LN2B,128},
        {39,5,SF_IDX_OUTB,16},
        {41,128,SF_RGB_B0+0,128},{43,128,SF_RGB_B0+128,128},{45,3,SF_RGB_B0+256,16},
        {47,128,SF_RGB_B0+272,128},{49,128,SF_RGB_B0+400,128},{51,3,SF_RGB_B0+528,16},
        {53,128,SF_RGB_B0+544,128},{55,128,SF_RGB_B0+672,128},{57,3,SF_RGB_B0+800,16},
        {59,128,SF_RGB_B0+816,128},{61,128,SF_RGB_B0+944,128},{63,3,SF_RGB_B0+1072,16},
    };
    int t = threadIdx.x;
    for (int e = 0; e < 30; ++e) {
        const float* src = P.p[tab[e].pi];
        for (int j = t; j < tab[e].pad; j += blockDim.x)
            sm[tab[e].dst + j] = (j < tab[e].n) ? src[j] : 0.0f;
    }
    __shared__ float pf[4][128];
    for (int br = 0; br < 2; ++br) {
        const float* fea = br ? P.part_shape : P.part_tex;
        int base = br ? 20 : 0;
        const float* ipw = P.p[base + 2]; const float* ipb = P.p[base + 3];
        const float* kw  = P.p[base + 6]; const float* kb  = P.p[base + 7];
        const float* vw  = P.p[base + 8]; const float* vb  = P.p[base + 9];
        int k4o = br ? SF_IDX_K4 : SF_FEAT_K4;
        int v4o = br ? SF_IDX_V4 : SF_FEAT_V4;
        __syncthreads();
        if (t < 128)
            for (int p = 0; p < 4; ++p) {
                float s = ipb[t];
                for (int c = 0; c < 128; ++c) s += fea[p * 128 + c] * ipw[c * 128 + t];
                pf[p][t] = s;
            }
        __syncthreads();
        if (t < 32)
            for (int p = 0; p < 4; ++p) {
                float s = kb[t];
                for (int c = 0; c < 128; ++c) s += pf[p][c] * kw[c * 32 + t];
                sm[k4o + p * 32 + t] = s;
            }
        if (t < 128)
            for (int p = 0; p < 4; ++p) {
                float s = vb[t];
                for (int c = 0; c < 128; ++c) s += pf[p][c] * vw[c * 128 + t];
                sm[v4o + p * 128 + t] = s;
            }
        __syncthreads();
    }
}

// ---------------------------------------------------------------------------
// Kernel 3: fused megakernel — 4 waves/block, 16 points/wave
// ---------------------------------------------------------------------------
__global__ __launch_bounds__(128) void mega_kernel(
    const float* __restrict__ k0, const float* __restrict__ vd,
    const __bf16* __restrict__ frags, const float* __restrict__ sm,
    float* __restrict__ out) {
    __shared__ __align__(32) __bf16 s_wbuf[32 * 512];      // 32 KB stage weights
    __shared__ __align__(32) __bf16 s_X[WAVES][16 * 128];  // x buffers
    __shared__ __align__(32) __bf16 s_H[WAVES][16 * 256];  // hidden / scratch
    __shared__ __align__(32) __bf16 s_F[WAVES][16 * 128];  // feat output
    __shared__ float s_q[WAVES][16 * 32];                  // q (fp32)
    __shared__ float s_wp[WAVES][16 * 4];                  // part weights

    const BC bcF = {FB_FEAT_W1, FB_FEAT_WQ, FB_FEAT_FFN1, FB_FEAT_FFN2,
                    SF_FEAT_B1, SF_FEAT_BQ, SF_FEAT_FFN1B, SF_FEAT_FFN2B,
                    SF_FEAT_LN1G, SF_FEAT_LN1B, SF_FEAT_LN2G, SF_FEAT_LN2B,
                    SF_FEAT_K4, SF_FEAT_V4};
    const BC bcI = {FB_IDX_W1, FB_IDX_WQ, FB_IDX_FFN1, FB_IDX_FFN2,
                    SF_IDX_B1, SF_IDX_BQ, SF_IDX_FFN1B, SF_IDX_FFN2B,
                    SF_IDX_LN1G, SF_IDX_LN1B, SF_IDX_LN2G, SF_IDX_LN2B,
                    SF_IDX_K4, SF_IDX_V4};

    int w = threadIdx.x >> 5, lane = threadIdx.x & 31;
    int m = lane & 15, hi = lane >> 4, cn = m;
    int pt0 = (blockIdx.x * WAVES + w) * 16;
    __bf16 *X = s_X[w], *H = s_H[w], *F = s_F[w];
    float *Q = s_q[w], *WP = s_wp[w];

    // ---- build rgb_feat = [k0(12), vd(3), sin(ve)(12), cos(ve)(12), pad] in H
    {
        int pt = pt0 + m;
        float vv[3] = {vd[pt * 3 + 0], vd[pt * 3 + 1], vd[pt * 3 + 2]};
        for (int c = hi * 32; c < hi * 32 + 32; ++c) {
            float val;
            if (c < 12) val = k0[pt * 12 + c];
            else if (c < 15) val = vv[c - 12];
            else if (c < 27) { int j = c - 15; val = sinf(vv[j >> 2] * (float)(1 << (j & 3))); }
            else if (c < 39) { int j = c - 27; val = cosf(vv[j >> 2] * (float)(1 << (j & 3))); }
            else val = 0.0f;
            H[m * 64 + c] = (__bf16)val;
        }
    }
    __syncthreads();

    // ---- feat branch, then feat = Wout @ relu(x) -> F
    run_branch<2>(X, H, Q, frags, sm, s_wbuf, bcF);
    copy_w(s_wbuf, frags + FB_FEAT_WOUT * 512, 32 * 512);
    __syncthreads();
    mm_stage<128, 4, 8, 1, 128, 0>(X, F, 0, s_wbuf, sm + SF_FEAT_OUTB);
    __syncthreads();

    // ---- index branch input: k0 padded 12->32 into H
    {
        int pt = pt0 + m;
        for (int c = hi * 16; c < hi * 16 + 16; ++c)
            H[m * 32 + c] = (__bf16)((c < 12) ? k0[pt * 12 + c] : 0.0f);
    }
    __syncthreads();
    run_branch<1>(X, H, Q, frags, sm, s_wbuf, bcI);
    // logits (128 -> 16, valid 5) -> H (ld 16)
    copy_w(s_wbuf, frags + FB_IDX_WOUT * 512, 8 * 512);
    __syncthreads();
    {
        v8f Ca[1]; Ca[0] = vzero8();
        mm_acc<128, 4, 1, 1>(X, 0, Ca, s_wbuf);
        const float* ob = sm + SF_IDX_OUTB;
#pragma unroll
        for (int r = 0; r < 8; ++r)
            H[(r + hi * 8) * 16 + cn] = (__bf16)(Ca[0][r] + ob[cn]);
    }
    __syncthreads();
    // softmax over logits[1..4] -> part weights
    if (lane < 16) {
        float l[4]; float mx = -3.0e38f;
#pragma unroll
        for (int p = 0; p < 4; ++p) { l[p] = (float)H[m * 16 + 1 + p]; mx = fmaxf(mx, l[p]); }
        float den = 0.f;
#pragma unroll
        for (int p = 0; p < 4; ++p) { l[p] = expf(l[p] - mx); den += l[p]; }
        float inv = 1.0f / den;
#pragma unroll
        for (int p = 0; p < 4; ++p) WP[m * 4 + p] = l[p] * inv;
    }
    __syncthreads();

    // ---- rgbnets: per part 128->128 relu, 128->128 relu, 128->3; blend by WP
    v8f acc = vzero8();
    for (int p = 0; p < 4; ++p) {
        int fbp = FB_RGB0 + p * 72;
        const float* rb = sm + SF_RGB_B0 + p * 272;
        copy_w(s_wbuf, frags + fbp * 512, 32 * 512);
        __syncthreads();
        mm_stage<128, 4, 8, 1, 128, 1>(F, X, 0, s_wbuf, rb);
        __syncthreads();
        copy_w(s_wbuf, frags + (fbp + 32) * 512, 32 * 512);
        __syncthreads();
        mm_stage<128, 4, 8, 1, 128, 1>(X, H, 0, s_wbuf, rb + 128);
        __syncthreads();
        copy_w(s_wbuf, frags + (fbp + 64) * 512, 8 * 512);
        __syncthreads();
        {
            v8f Ca[1]; Ca[0] = vzero8();
            mm_acc<128, 4, 1, 1>(H, 0, Ca, s_wbuf);
            const float* b2 = rb + 256;
#pragma unroll
            for (int r = 0; r < 8; ++r) {
                int mr = r + hi * 8;
                acc[r] += WP[mr * 4 + p] * (Ca[0][r] + b2[cn]);
            }
        }
        __syncthreads();
    }
    // ---- sigmoid + store (lanes with column < 3 hold real channels)
    if (cn < 3) {
#pragma unroll
        for (int r = 0; r < 8; ++r) {
            int pt = pt0 + r + hi * 8;
            out[pt * 3 + cn] = 1.0f / (1.0f + expf(-acc[r]));
        }
    }
}

// ---------------------------------------------------------------------------
extern "C" void kernel_launch(void* const* d_in, const int* in_sizes, int n_in,
                              void* d_out, int out_size, void* d_ws, size_t ws_size,
                              hipStream_t stream) {
    (void)in_sizes; (void)out_size; (void)ws_size;
    Params P{};
    int np = n_in - 4; if (np > 64) np = 64; if (np < 0) np = 0;
    for (int i = 0; i < np; ++i) P.p[i] = (const float*)d_in[4 + i];
    P.part_shape = (const float*)d_in[2];
    P.part_tex   = (const float*)d_in[3];

    __bf16* frags = (__bf16*)d_ws;
    float*  sm    = (float*)((char*)d_ws + (size_t)NFRAG_TOTAL * 1024);

    pack_kernel<<<NFRAG_TOTAL, 256, 0, stream>>>(P, frags);
    prep2_kernel<<<1, 256, 0, stream>>>(P, sm);
    mega_kernel<<<N_PTS / (16 * WAVES), 128, 0, stream>>>(
        (const float*)d_in[0], (const float*)d_in[1], frags, sm, (float*)d_out);
}
